// CrossAttentionFusion_67912022884693
// MI455X (gfx1250) — matrix operand order repaired and checked by reading, not compiled, not run on hardware
//
#include <hip/hip_runtime.h>

// ---------------------------------------------------------------------------
// CrossAttentionFusion for MI455X (gfx1250, wave32, WMMA 16x16x32 bf16)
//   Q = x1@Wq+bq ; K = x2@Wk+bk ; V = x2@Wv+bv
//   out = softmax(Q K^T) V + x1
// Roofline: ~49 GFLOP vs ~84 MB I/O -> run matmuls on bf16 WMMA, f32 accum.
// Scratch layout in d_ws (bf16): Q[B*SQ*DH], K[B*SK*DH], V^T[B*DH*SK] = 24 MB.
// ---------------------------------------------------------------------------

#define B_   8
#define SQ_  2048
#define SK_  2048
#define D1_  256
#define D2_  768
#define DH_  256
#define SLOG 11          // log2(SQ_) == log2(SK_)

typedef __attribute__((ext_vector_type(16))) __bf16 v16bf;
typedef __attribute__((ext_vector_type(8)))  float  v8f;

union Frag16 {               // one WMMA 16-bit A/B operand (8 dwords / lane)
  v16bf        v;
  unsigned int d[8];
};

static __device__ __forceinline__ unsigned short f2bf(float f) {
  unsigned int u = __float_as_uint(f);
  u = (u + 0x7FFFu + ((u >> 16) & 1u)) >> 16;          // round-to-nearest-even
  return (unsigned short)u;
}

static __device__ __forceinline__ unsigned int pack2(float lo, float hi) {
#if __has_builtin(__builtin_amdgcn_cvt_pk_bf16_f32)
  typedef __attribute__((ext_vector_type(2))) __bf16 v2bf;
  v2bf p = __builtin_amdgcn_cvt_pk_bf16_f32(lo, hi);   // v_cvt_pk_bf16_f32
  union { v2bf v; unsigned int u; } cvt;
  cvt.v = p;
  return cvt.u;
#else
  return ((unsigned int)f2bf(hi) << 16) | (unsigned int)f2bf(lo);
#endif
}

// A-operand (16-bit, 16x32): dword r of lane(half h) holds K = k0, k0+1
static __device__ __forceinline__ int a_k0(int r, int h) {
  return ((r < 4) ? 2 * r : 16 + 2 * (r - 4)) + 8 * h;
}
// B-operand (16-bit, 32x16): dword r of lane(half h) holds K = kd, kd+1
static __device__ __forceinline__ int b_kd(int r, int h) {
  return ((r < 4) ? 2 * r : 16 + 2 * (r - 4)) + 8 * h;
}

// ---------------------------------------------------------------------------
// proj_kernel: out[M x 256] = X[M x Kd] @ W[Kd x 256] + bias  (bf16 out)
// Block: 256 threads = 8 waves. Block tile = 16 rows x 128 cols,
// wave w owns cols [n0, n0+16). TRANS=1 stores per-batch [256 x 2048].
// X tile staged f32 in LDS via CDNA5 async global->LDS copies.
// ---------------------------------------------------------------------------
template <int TRANS>
__global__ void __launch_bounds__(256)
proj_kernel(const float* __restrict__ X, const float* __restrict__ W,
            const float* __restrict__ bias, unsigned short* __restrict__ out,
            int Kd) {
  __shared__ float xt[16 * 32];                 // X tile, f32 (2 KB)

  const int tid  = threadIdx.x;
  const int w    = tid >> 5;
  const int lane = tid & 31;
  const int h    = lane >> 4;
  const int ml   = lane & 15;

  const int r0 = blockIdx.x * 16;               // row base
  const int n0 = blockIdx.y * 128 + w * 16;     // col base for this wave

  const unsigned lds_base = (unsigned)(size_t)(&xt[0]);  // LDS offset bits

  v8f acc;
  #pragma unroll
  for (int i = 0; i < 8; ++i) acc[i] = 0.0f;

  for (int kk = 0; kk < Kd; kk += 32) {
    // async-stage X[16][32] f32 into LDS: 8 bytes per thread
    {
      const int e  = tid * 2;                   // 512 floats total
      const int rr = e >> 5, cc = e & 31;
      const float*   src   = X + (size_t)(r0 + rr) * Kd + kk + cc;
      const unsigned ldsb  = lds_base + (unsigned)e * 4u;
      asm volatile("global_load_async_to_lds_b64 %0, %1, off"
                   :: "v"(ldsb), "v"(src) : "memory");
    }
    asm volatile("s_wait_asynccnt 0x0" ::: "memory");
    __syncthreads();

    Frag16 af;                                  // A from LDS (convert to bf16)
    #pragma unroll
    for (int r = 0; r < 8; ++r) {
      const int k0 = a_k0(r, h);
      af.d[r] = pack2(xt[ml * 32 + k0], xt[ml * 32 + k0 + 1]);
    }

    Frag16 bfrag;                               // B from W (pairs strided 256)
    #pragma unroll
    for (int r = 0; r < 8; ++r) {
      const int kd = b_kd(r, h);
      const float w0 = W[(size_t)(kk + kd)     * 256 + n0 + ml];
      const float w1 = W[(size_t)(kk + kd + 1) * 256 + n0 + ml];
      bfrag.d[r] = pack2(w0, w1);
    }

    acc = __builtin_amdgcn_wmma_f32_16x16x32_bf16(
        false, af.v, false, bfrag.v, (short)0, acc, false, false);
    __syncthreads();
  }

  const float bval = bias[n0 + ml];
  #pragma unroll
  for (int r = 0; r < 8; ++r) {
    const int row = r0 + r + 8 * h;             // C layout: row = r + 8*half
    const int col = n0 + ml;
    const unsigned short ob = f2bf(acc[r] + bval);
    if (TRANS == 0) {
      out[(size_t)row * 256 + col] = ob;
    } else {
      const int bb = row >> SLOG;               // S == 2048 (compile-time)
      const int ss = row & (SK_ - 1);
      out[((size_t)bb * 256 + col) * SK_ + ss] = ob;
    }
  }
}

// ---------------------------------------------------------------------------
// attn_kernel: flash attention + residual. 128 threads = 4 waves per block,
// each wave owns 16 query rows. Keys processed 32 at a time (two 16-wide
// score tiles -> one 16x32 bf16 P A-fragment via per-wave LDS relayout).
// ---------------------------------------------------------------------------
__global__ void __launch_bounds__(128)
attn_kernel(const unsigned short* __restrict__ Qbf,
            const unsigned short* __restrict__ Kbf,
            const unsigned short* __restrict__ VT,   // [B][DH][SK] bf16
            const float* __restrict__ x1, float* __restrict__ out) {
  __shared__ unsigned short lds_p[4][16 * 32];

  const int tid  = threadIdx.x;
  const int w    = tid >> 5;
  const int lane = tid & 31;
  const int h    = lane >> 4;
  const int ml   = lane & 15;

  const int tile = blockIdx.x * 4 + w;          // global 16-query tile id
  const int b    = tile >> (SLOG - 4);          // tile / (SQ_/16)
  const int q0   = (tile & (SQ_ / 16 - 1)) * 16;

  // resident Q fragments: 16 x 256 bf16 -> 8 A-fragments (64 VGPRs)
  Frag16 qf[8];
  {
    const unsigned short* Qrow = Qbf + (size_t)(b * SQ_ + q0 + ml) * DH_;
    #pragma unroll
    for (int f = 0; f < 8; ++f)
      #pragma unroll
      for (int r = 0; r < 8; ++r)
        qf[f].d[r] = *(const unsigned int*)(Qrow + f * 32 + a_k0(r, h));
  }

  v8f acc[16];                                  // O: 16 x 256 f32
  #pragma unroll
  for (int f = 0; f < 16; ++f)
    #pragma unroll
    for (int r = 0; r < 8; ++r) acc[f][r] = 0.0f;

  float mrow[8], lrow[8];                       // stats for row = r + 8*half
  #pragma unroll
  for (int r = 0; r < 8; ++r) { mrow[r] = -1e30f; lrow[r] = 0.0f; }

  const unsigned short* Vb = VT + (size_t)b * DH_ * SK_;
  unsigned short* pw = &lds_p[w][0];

  for (int kb = 0; kb < SK_ / 32; ++kb) {
    const int kbase = kb * 32;

    if (kb + 1 < SK_ / 32) {                    // -> global_prefetch_b8
      __builtin_prefetch(Kbf + (size_t)(b * SK_ + kbase + 32 + ml) * DH_, 0, 3);
      __builtin_prefetch(Vb + (size_t)(ml * 16) * SK_ + kbase + 32, 0, 3);
    }

    // ---- S = Q K^T for 32 keys (two 16x16 f32 tiles) ----
    v8f s0, s1;
    #pragma unroll
    for (int r = 0; r < 8; ++r) { s0[r] = 0.0f; s1[r] = 0.0f; }

    const unsigned short* Kt0 = Kbf + (size_t)(b * SK_ + kbase + ml) * DH_;
    const unsigned short* Kt1 = Kt0 + (size_t)16 * DH_;
    #pragma unroll
    for (int f = 0; f < 8; ++f) {
      Frag16 kf0, kf1;
      #pragma unroll
      for (int r = 0; r < 8; ++r) {
        const int kd = b_kd(r, h);
        kf0.d[r] = *(const unsigned int*)(Kt0 + f * 32 + kd);
        kf1.d[r] = *(const unsigned int*)(Kt1 + f * 32 + kd);
      }
      s0 = __builtin_amdgcn_wmma_f32_16x16x32_bf16(
          false, qf[f].v, false, kf0.v, (short)0, s0, false, false);
      s1 = __builtin_amdgcn_wmma_f32_16x16x32_bf16(
          false, qf[f].v, false, kf1.v, (short)0, s1, false, false);
    }

    // ---- online softmax: row reductions across the 16 lanes of a half ----
    float alpha[8];
    #pragma unroll
    for (int r = 0; r < 8; ++r) {
      float mx = fmaxf(s0[r], s1[r]);
      mx = fmaxf(mx, __shfl_xor(mx, 1, 32));
      mx = fmaxf(mx, __shfl_xor(mx, 2, 32));
      mx = fmaxf(mx, __shfl_xor(mx, 4, 32));
      mx = fmaxf(mx, __shfl_xor(mx, 8, 32));
      const float mnew = fmaxf(mrow[r], mx);
      alpha[r] = __expf(mrow[r] - mnew);
      mrow[r]  = mnew;
      const float p0 = __expf(s0[r] - mnew);
      const float p1 = __expf(s1[r] - mnew);
      s0[r] = p0; s1[r] = p1;
      float rs = p0 + p1;
      rs += __shfl_xor(rs, 1, 32);
      rs += __shfl_xor(rs, 2, 32);
      rs += __shfl_xor(rs, 4, 32);
      rs += __shfl_xor(rs, 8, 32);
      lrow[r] = lrow[r] * alpha[r] + rs;
    }
    #pragma unroll
    for (int f = 0; f < 16; ++f)
      #pragma unroll
      for (int r = 0; r < 8; ++r) acc[f][r] *= alpha[r];

    // ---- relayout P (C layout -> A layout) through per-wave LDS tile ----
    #pragma unroll
    for (int r = 0; r < 8; ++r) {
      const int row = r + 8 * h;
      pw[row * 32 + ml]      = f2bf(s0[r]);
      pw[row * 32 + 16 + ml] = f2bf(s1[r]);
    }
    Frag16 pf;                                  // same-wave DS ops are in-order
    #pragma unroll
    for (int r = 0; r < 8; ++r)
      pf.d[r] = *(const unsigned int*)(pw + ml * 32 + a_k0(r, h));

    // ---- O += P V (V^T layout: key pairs contiguous) ----
    #pragma unroll
    for (int f = 0; f < 16; ++f) {
      Frag16 vf;
      #pragma unroll
      for (int r = 0; r < 8; ++r) {
        const int kd = b_kd(r, h);
        vf.d[r] = *(const unsigned int*)(Vb + (size_t)(f * 16 + ml) * SK_ +
                                         kbase + kd);
      }
      acc[f] = __builtin_amdgcn_wmma_f32_16x16x32_bf16(
          false, pf.v, false, vf.v, (short)0, acc[f], false, false);
    }
  }

  // ---- epilogue: O / l + residual ----
  #pragma unroll
  for (int f = 0; f < 16; ++f) {
    #pragma unroll
    for (int r = 0; r < 8; ++r) {
      const int row = q0 + r + 8 * h;
      const int col = f * 16 + ml;
      const size_t idx = (size_t)(b * SQ_ + row) * DH_ + col;
      out[idx] = acc[f][r] / lrow[r] + x1[idx];
    }
  }
}

// ---------------------------------------------------------------------------
extern "C" void kernel_launch(void* const* d_in, const int* in_sizes, int n_in,
                              void* d_out, int out_size, void* d_ws,
                              size_t ws_size, hipStream_t stream) {
  (void)in_sizes; (void)n_in; (void)out_size; (void)ws_size;
  const float* x1 = (const float*)d_in[0];
  const float* x2 = (const float*)d_in[1];
  const float* Wq = (const float*)d_in[2];
  const float* bq = (const float*)d_in[3];
  const float* Wk = (const float*)d_in[4];
  const float* bk = (const float*)d_in[5];
  const float* Wv = (const float*)d_in[6];
  const float* bv = (const float*)d_in[7];
  float* out = (float*)d_out;

  unsigned short* Qbf = (unsigned short*)d_ws;                // B*SQ*DH bf16
  unsigned short* Kbf = Qbf + (size_t)B_ * SQ_ * DH_;         // B*SK*DH bf16
  unsigned short* VT  = Kbf + (size_t)B_ * SK_ * DH_;         // B*DH*SK bf16

  const dim3 blk(256);
  proj_kernel<0><<<dim3(B_ * SQ_ / 16, DH_ / 128), blk, 0, stream>>>(
      x1, Wq, bq, Qbf, D1_);
  proj_kernel<0><<<dim3(B_ * SK_ / 16, DH_ / 128), blk, 0, stream>>>(
      x2, Wk, bk, Kbf, D2_);
  proj_kernel<1><<<dim3(B_ * SK_ / 16, DH_ / 128), blk, 0, stream>>>(
      x2, Wv, bv, VT, D2_);

  attn_kernel<<<dim3(B_ * SQ_ / 64), dim3(128), 0, stream>>>(
      Qbf, Kbf, VT, x1, out);
}
